// cGRU_59760174956700
// MI455X (gfx1250) — matrix operand order repaired and checked by reading, not compiled, not run on hardware
//
#include <hip/hip_runtime.h>
#include <math.h>

// ---------------------------------------------------------------------------
// cGRU on MI455X (gfx1250): bf16 WMMA GEMMs + fp32 state update.
//   B=64, T=512, D=H=1024
//   Phase 1: Xz/Xr/Xh = x @ Wf*.T + b   (big parallel GEMM, wmma bf16)
//   Phase 2: 512 sequential steps, 2 kernels/step (gates, update)
// Weight tiles are staged in LDS (double-buffered, async-to-LDS via
// GLOBAL_LOAD_ASYNC_TO_LDS_B128 / ASYNCcnt), A fragments register-pipelined.
// ---------------------------------------------------------------------------

#define GB   64
#define GT   512
#define GD   1024
#define GH   1024
#define GMIN (GB * GT)   // 32768 rows for the input projection GEMM

typedef __attribute__((ext_vector_type(16))) __bf16 v16bf;
typedef __attribute__((ext_vector_type(8)))  __bf16 v8bf;
typedef __attribute__((ext_vector_type(8)))  float  v8f;
typedef __attribute__((__vector_size__(16))) int    v4i;   // 16-byte chunk type

#if __has_builtin(__builtin_amdgcn_global_load_async_to_lds_b128) && \
    __has_builtin(__builtin_amdgcn_s_wait_asynccnt)
#define USE_ASYNC_LDS 1
#else
#define USE_ASYNC_LDS 0
#endif

#define AS1 __attribute__((address_space(1)))
#define AS3 __attribute__((address_space(3)))

__device__ __forceinline__ v16bf cat8(v8bf lo, v8bf hi) {
    return __builtin_shufflevector(lo, hi, 0,1,2,3,4,5,6,7,8,9,10,11,12,13,14,15);
}

// 16 bytes global -> LDS. Async (ASYNCcnt-tracked DMA) when available.
__device__ __forceinline__ void cp_b128(__bf16* ldst, const __bf16* gsrc) {
#if USE_ASYNC_LDS
    __builtin_amdgcn_global_load_async_to_lds_b128(
        (AS1 v4i*)(unsigned long long)gsrc,
        (AS3 v4i*)(unsigned int)(unsigned long long)ldst,
        0, 0);
#else
    *(v8bf*)ldst = *(const v8bf*)gsrc;
#endif
}

__device__ __forceinline__ void wait_async_le2() {
#if USE_ASYNC_LDS
    __builtin_amdgcn_s_wait_asynccnt(2);
#endif
}
__device__ __forceinline__ void wait_async_le0() {
#if USE_ASYNC_LDS
    __builtin_amdgcn_s_wait_asynccnt(0);
#endif
}

// A fragment: 16x32 (MxK) bf16, CDNA5 layout:
//   lanes 0-15 rows m, VGPR0-3 => K 0..7,  VGPR4-7 => K 16..23
//   lanes 16-31 same rows,     VGPR0-3 => K 8..15, VGPR4-7 => K 24..31
__device__ __forceinline__ v16bf load_a_frag(const __bf16* __restrict__ A, int lda,
                                             int m_base, int k0, int lane) {
    const int m  = m_base + (lane & 15);
    const int hi = (lane >> 4) & 1;
    const __bf16* p = A + (size_t)m * lda + (k0 + hi * 8);
    v8bf x0 = *(const v8bf*)(p);        // K = k0 + hi*8      .. +7
    v8bf x1 = *(const v8bf*)(p + 16);   // K = k0 + 16 + hi*8 .. +7
    return cat8(x0, x1);
}

// B fragment from the LDS-staged chunk [64 rows(n-local) x 32 k], row pitch 32:
//   lane: n = lane&15 ; lanes 0-15 K 0..7,16..23, lanes 16-31 K 8..15,24..31
__device__ __forceinline__ v16bf lds_b_frag(const __bf16* ldsB, int n_local, int lane) {
    const int n  = n_local + (lane & 15);
    const int hi = (lane >> 4) & 1;
    const __bf16* p = ldsB + n * 32 + hi * 8;
    v8bf x0 = *(const v8bf*)(p);
    v8bf x1 = *(const v8bf*)(p + 16);
    return cat8(x0, x1);
}

__device__ __forceinline__ v8f wmma_bf16(v16bf a, v16bf b, v8f c) {
    // (neg_a, A, neg_b, B, c_mod, C, reuse_a, reuse_b)
    return __builtin_amdgcn_wmma_f32_16x16x32_bf16(false, a, false, b, (short)0, c,
                                                   false, false);
}

__device__ __forceinline__ float sigm(float x) { return 1.0f / (1.0f + __expf(-x)); }

// ---------------------------------------------------------------------------
// Core tile routine: acc[4] += A[m_base:+16, :K] @ W[n_base:+64, :K]^T
// A 4-wave group (gtid 0..127) cooperates on an LDS double-buffer for W.
// ldsB points to this group's region: [2 buffers][64*32] bf16 (8 KB).
// ---------------------------------------------------------------------------
__device__ __forceinline__ void gemm_tile_64x64(
    const __bf16* __restrict__ A, int lda,
    const __bf16* __restrict__ W, int ldw, int K,
    int m_base, int n_base, int gtid,
    __bf16* ldsB, v8f acc[4]) {

    const int lane  = gtid & 31;
    const int crow  = gtid >> 1;     // 0..63: n-row this thread stages
    const int chalf = gtid & 1;      // which 16-element half of the 32-k chunk
    const __bf16* gsrc = W + (size_t)(n_base + crow) * ldw + chalf * 16;
    __bf16* lrow = ldsB + crow * 32 + chalf * 16;

    const int nk = K / 32;

    // prefill buffer 0 with k-chunk 0  (2 x b128 per thread)
    cp_b128(lrow,     gsrc);
    cp_b128(lrow + 8, gsrc + 8);

    v16bf a_cur = load_a_frag(A, lda, m_base, 0, lane);

    for (int kc = 0; kc < nk; ++kc) {
        const int cur = kc & 1;
        if (kc + 1 < nk) {
            // stage next chunk into the other buffer, then wait for current
            const __bf16* gn = gsrc + (kc + 1) * 32;
            __bf16*       ln = ldsB + (cur ^ 1) * (64 * 32) + crow * 32 + chalf * 16;
            cp_b128(ln,     gn);
            cp_b128(ln + 8, gn + 8);
            wait_async_le2();        // my 2 current-buffer copies have landed
        } else {
            wait_async_le0();
        }
        __syncthreads();             // whole group's current buffer visible

        // register-pipeline the A fragment for the next chunk
        v16bf a_nxt = a_cur;
        if (kc + 1 < nk) {
            a_nxt = load_a_frag(A, lda, m_base, (kc + 1) * 32, lane);
            if (kc + 2 < nk)
                __builtin_prefetch(A + (size_t)(m_base + (lane & 15)) * lda +
                                   (kc + 2) * 32, 0, 1);
        }

        const __bf16* bcur = ldsB + cur * (64 * 32);
#pragma unroll
        for (int i = 0; i < 4; ++i) {
            v16bf b = lds_b_frag(bcur, i * 16, lane);
            acc[i] = wmma_bf16(a_cur, b, acc[i]);
        }
        a_cur = a_nxt;
        __syncthreads();             // done reading 'cur' before it is refilled
    }
}

// ---------------------------------------------------------------------------
// fp32 -> bf16 conversion (grid-stride)
// ---------------------------------------------------------------------------
__global__ void k_f32_to_bf16(const float* __restrict__ in, __bf16* __restrict__ out,
                              long long n) {
    long long i = (long long)blockIdx.x * blockDim.x + threadIdx.x;
    long long stride = (long long)gridDim.x * blockDim.x;
    for (; i < n; i += stride) out[i] = (__bf16)in[i];
}

// ---------------------------------------------------------------------------
// Input projections: O[z|r|h] = bf16( x @ W.T + bias )
// grid (GMIN/64, GH/64, 3), block 128 (4 waves); wave => 16x64 tile
// ---------------------------------------------------------------------------
__global__ void k_gemm_in(const __bf16* __restrict__ X,
                          const __bf16* __restrict__ W0, const __bf16* __restrict__ W1,
                          const __bf16* __restrict__ W2,
                          const float* __restrict__ b0, const float* __restrict__ b1,
                          const float* __restrict__ b2,
                          __bf16* __restrict__ O0, __bf16* __restrict__ O1,
                          __bf16* __restrict__ O2) {
    __shared__ __bf16 ldsB[2 * 64 * 32];
    const int lane   = threadIdx.x & 31;
    const int wave   = threadIdx.x >> 5;
    const int m_base = blockIdx.x * 64 + wave * 16;
    const int n_base = blockIdx.y * 64;

    const __bf16* W   = (blockIdx.z == 0) ? W0 : (blockIdx.z == 1) ? W1 : W2;
    const float*  bia = (blockIdx.z == 0) ? b0 : (blockIdx.z == 1) ? b1 : b2;
    __bf16*       O   = (blockIdx.z == 0) ? O0 : (blockIdx.z == 1) ? O1 : O2;

    v8f acc[4] = {};
    gemm_tile_64x64(X, GD, W, GD, GD, m_base, n_base, threadIdx.x, ldsB, acc);

    const int col = lane & 15, rs = (lane >> 4) & 1;
#pragma unroll
    for (int i = 0; i < 4; ++i) {
        const int n  = n_base + i * 16 + col;
        const float bv = bia[n];
#pragma unroll
        for (int v = 0; v < 8; ++v) {
            const int m = m_base + rs * 8 + v;
            O[(size_t)m * GH + n] = (__bf16)(acc[i][v] + bv);
        }
    }
}

// ---------------------------------------------------------------------------
// Step kernel A: z = sigmoid(Xz[t] + h@Wuz.T + buz)  (waves 0-3)
//                r = sigmoid(Xr[t] + h@Wur.T + bur); rh = r*h (waves 4-7)
// grid (GH/64), block 256 (8 waves, two 4-wave groups)
// ---------------------------------------------------------------------------
__global__ void k_step_gates(const __bf16* __restrict__ h_b, const float* __restrict__ h_f,
                             const __bf16* __restrict__ Wuz, const __bf16* __restrict__ Wur,
                             const float* __restrict__ buz, const float* __restrict__ bur,
                             const __bf16* __restrict__ Xz, const __bf16* __restrict__ Xr,
                             int t, float* __restrict__ z_f, __bf16* __restrict__ rh_b) {
    __shared__ __bf16 ldsB[2][2 * 64 * 32];   // one double-buffer per group
    const int lane   = threadIdx.x & 31;
    const int wave   = threadIdx.x >> 5;
    const int group  = wave >> 2;             // 0 => z path, 1 => r path
    const int gtid   = threadIdx.x & 127;
    const int m_base = (wave & 3) * 16;
    const int n_base = blockIdx.x * 64;

    const __bf16* W  = group ? Wur : Wuz;
    const __bf16* Xg = group ? Xr  : Xz;
    const float* bia = group ? bur : buz;

    v8f acc[4] = {};
    gemm_tile_64x64(h_b, GH, W, GH, GH, m_base, n_base, gtid, ldsB[group], acc);

    const int col = lane & 15, rs = (lane >> 4) & 1;
#pragma unroll
    for (int i = 0; i < 4; ++i) {
        const int n  = n_base + i * 16 + col;
        const float bv = bia[n];
#pragma unroll
        for (int v = 0; v < 8; ++v) {
            const int m = m_base + rs * 8 + v;
            const float xg = (float)Xg[((size_t)m * GT + t) * GH + n];
            const float g  = sigm(acc[i][v] + xg + bv);
            const size_t idx = (size_t)m * GH + n;
            if (group) rh_b[idx] = (__bf16)(g * h_f[idx]);
            else       z_f[idx]  = g;
        }
    }
}

// ---------------------------------------------------------------------------
// Step kernel B: hh = silu(Xh[t] + rh@Wuh.T + buh); h = (1-z)h + z*hh
// grid (GH/64), block 128 (4 waves)
// ---------------------------------------------------------------------------
__global__ void k_step_update(const __bf16* __restrict__ rh_b,
                              const __bf16* __restrict__ Wuh,
                              const float* __restrict__ buh,
                              const __bf16* __restrict__ Xh,
                              const float* __restrict__ z_f,
                              float* __restrict__ h_f, __bf16* __restrict__ h_b,
                              float* __restrict__ y, int t) {
    __shared__ __bf16 ldsB[2 * 64 * 32];
    const int lane   = threadIdx.x & 31;
    const int wave   = threadIdx.x >> 5;
    const int m_base = wave * 16;
    const int n_base = blockIdx.x * 64;

    v8f acc[4] = {};
    gemm_tile_64x64(rh_b, GH, Wuh, GH, GH, m_base, n_base, threadIdx.x, ldsB, acc);

    const int col = lane & 15, rs = (lane >> 4) & 1;
#pragma unroll
    for (int i = 0; i < 4; ++i) {
        const int n  = n_base + i * 16 + col;
        const float bv = buh[n];
#pragma unroll
        for (int v = 0; v < 8; ++v) {
            const int m = m_base + rs * 8 + v;
            const float s  = acc[i][v] + (float)Xh[((size_t)m * GT + t) * GH + n] + bv;
            const float hh = s * sigm(s);                       // silu
            const size_t idx = (size_t)m * GH + n;
            const float zz = z_f[idx];
            const float hn = (1.0f - zz) * h_f[idx] + zz * hh;
            y[((size_t)m * GT + t) * GH + n] = hn;
            h_f[idx] = hn;
            h_b[idx] = (__bf16)hn;
        }
    }
}

// ---------------------------------------------------------------------------
__global__ void k_init_h(const float* __restrict__ h0, float* __restrict__ h_f,
                         __bf16* __restrict__ h_b, int n) {
    int i = blockIdx.x * blockDim.x + threadIdx.x;
    if (i < n) { h_f[i] = h0[i]; h_b[i] = (__bf16)h0[i]; }
}

__global__ void k_copy_f32(const float* __restrict__ src, float* __restrict__ dst, int n) {
    int i = blockIdx.x * blockDim.x + threadIdx.x;
    if (i < n) dst[i] = src[i];
}

// ---------------------------------------------------------------------------
extern "C" void kernel_launch(void* const* d_in, const int* in_sizes, int n_in,
                              void* d_out, int out_size, void* d_ws, size_t ws_size,
                              hipStream_t stream) {
    (void)in_sizes; (void)n_in; (void)out_size; (void)ws_size;
    const float* x   = (const float*)d_in[0];
    const float* h0  = (const float*)d_in[1];
    const float* Wfz = (const float*)d_in[2];  const float* bfz = (const float*)d_in[3];
    const float* Wfr = (const float*)d_in[4];  const float* bfr = (const float*)d_in[5];
    const float* Wfh = (const float*)d_in[6];  const float* bfh = (const float*)d_in[7];
    const float* Wuz = (const float*)d_in[8];  const float* buz = (const float*)d_in[9];
    const float* Wur = (const float*)d_in[10]; const float* bur = (const float*)d_in[11];
    const float* Wuh = (const float*)d_in[12]; const float* buh = (const float*)d_in[13];

    float* y      = (float*)d_out;                       // [B,T,H]
    float* h_last = y + (size_t)GB * GT * GH;            // [B,1,H]

    // ---- workspace carve-up (bf16 operands + fp32 state) -------------------
    char* w = (char*)d_ws;
    auto take = [&](size_t bytes) -> char* {
        char* p = w; w += (bytes + 255) & ~(size_t)255; return p;
    };
    __bf16* x_b    = (__bf16*)take((size_t)GMIN * GD * 2);   // 64 MB
    __bf16* Wfz_b  = (__bf16*)take((size_t)GH * GD * 2);
    __bf16* Wfr_b  = (__bf16*)take((size_t)GH * GD * 2);
    __bf16* Wfh_b  = (__bf16*)take((size_t)GH * GD * 2);
    __bf16* Wuz_b  = (__bf16*)take((size_t)GH * GH * 2);
    __bf16* Wur_b  = (__bf16*)take((size_t)GH * GH * 2);
    __bf16* Wuh_b  = (__bf16*)take((size_t)GH * GH * 2);    // weights: 12 MB, L2-resident
    __bf16* Xz_b   = (__bf16*)take((size_t)GMIN * GH * 2);  // 64 MB
    __bf16* Xr_b   = (__bf16*)take((size_t)GMIN * GH * 2);
    __bf16* Xh_b   = (__bf16*)take((size_t)GMIN * GH * 2);
    float*  h_f    = (float*) take((size_t)GB * GH * 4);
    __bf16* h_b    = (__bf16*)take((size_t)GB * GH * 2);
    float*  z_f    = (float*) take((size_t)GB * GH * 4);
    __bf16* rh_b   = (__bf16*)take((size_t)GB * GH * 2);

    // ---- phase 0: precision conversion ------------------------------------
    k_f32_to_bf16<<<2048, 256, 0, stream>>>(x,   x_b,   (long long)GMIN * GD);
    k_f32_to_bf16<<<1024, 256, 0, stream>>>(Wfz, Wfz_b, (long long)GH * GD);
    k_f32_to_bf16<<<1024, 256, 0, stream>>>(Wfr, Wfr_b, (long long)GH * GD);
    k_f32_to_bf16<<<1024, 256, 0, stream>>>(Wfh, Wfh_b, (long long)GH * GD);
    k_f32_to_bf16<<<1024, 256, 0, stream>>>(Wuz, Wuz_b, (long long)GH * GH);
    k_f32_to_bf16<<<1024, 256, 0, stream>>>(Wur, Wur_b, (long long)GH * GH);
    k_f32_to_bf16<<<1024, 256, 0, stream>>>(Wuh, Wuh_b, (long long)GH * GH);

    // ---- phase 1: batched input projections (3 GEMMs via grid.z) ----------
    dim3 gIn(GMIN / 64, GH / 64, 3);
    k_gemm_in<<<gIn, 128, 0, stream>>>(x_b, Wfz_b, Wfr_b, Wfh_b, bfz, bfr, bfh,
                                       Xz_b, Xr_b, Xh_b);

    // ---- phase 2: sequential scan ------------------------------------------
    k_init_h<<<(GB * GH + 255) / 256, 256, 0, stream>>>(h0, h_f, h_b, GB * GH);
    for (int t = 0; t < GT; ++t) {
        k_step_gates<<<dim3(GH / 64), 256, 0, stream>>>(h_b, h_f, Wuz_b, Wur_b,
                                                        buz, bur, Xz_b, Xr_b, t,
                                                        z_f, rh_b);
        k_step_update<<<dim3(GH / 64), 128, 0, stream>>>(rh_b, Wuh_b, buh, Xh_b,
                                                         z_f, h_f, h_b, y, t);
    }
    k_copy_f32<<<(GB * GH + 255) / 256, 256, 0, stream>>>(h_f, h_last, GB * GH);
}
